// Vanilla_att_10668698763871
// MI455X (gfx1250) — compile-verified, matrix operand order checked by reading, not run on hardware
//
#include <hip/hip_runtime.h>
#include <math.h>

#define B_   64
#define T_   1024
#define V_   50000
#define D_   300
#define DP_  320
#define H_   256
#define HD_  128
#define G4_  512   // 4*HD
#define ATT_ 128
#define CAT_ 5
#define BG_  16    // batches per scan workgroup

typedef __bf16 bf16_t;
typedef bf16_t v16bf __attribute__((ext_vector_type(16)));
typedef bf16_t v8bf  __attribute__((ext_vector_type(8)));
typedef float  v8f   __attribute__((ext_vector_type(8)));

__device__ __forceinline__ bf16_t f2bf(float f) {
  unsigned u = __builtin_bit_cast(unsigned, f);
  unsigned r = (u + 0x7FFFu + ((u >> 16) & 1u)) >> 16;
  unsigned short s = (unsigned short)r;
  return __builtin_bit_cast(bf16_t, s);
}
__device__ __forceinline__ float bf2f(bf16_t b) {
  unsigned short s = __builtin_bit_cast(unsigned short, b);
  unsigned u = ((unsigned)s) << 16;
  return __builtin_bit_cast(float, u);
}
__device__ __forceinline__ float sigmoidf_(float x) { return 1.0f / (1.0f + __expf(-x)); }

// ---- CDNA5 async global->LDS (ASYNCcnt) ----------------------------------
__device__ __forceinline__ unsigned lds_off(const void* p) {
  return (unsigned)(unsigned long long)(size_t)p;  // flat LDS addr -> LDS offset
}
__device__ __forceinline__ void async_ld_b128(unsigned ldsoff, const void* g) {
  asm volatile("global_load_async_to_lds_b128 %0, %1, off"
               :: "v"(ldsoff), "v"(g) : "memory");
}
__device__ __forceinline__ void wait_async0() {
  asm volatile("s_wait_asynccnt 0" ::: "memory");
}
__device__ __forceinline__ void wait_async3() {
  asm volatile("s_wait_asynccnt 3" ::: "memory");  // 3 newest (next stage) may remain
}

// ---- WMMA fragment helpers (CDNA5 16x16x32 bf16 layouts) ------------------
__device__ __forceinline__ v16bf frag_ab(const bf16_t* base, int stride) {
  const int lane = threadIdx.x & 31;
  const bf16_t* p = base + (lane & 15) * stride + ((lane >> 4) << 3);
  v8bf lo = *(const v8bf*)p;
  v8bf hi = *(const v8bf*)(p + 16);
  return __builtin_shufflevector(lo, hi, 0,1,2,3,4,5,6,7,8,9,10,11,12,13,14,15);
}
__device__ __forceinline__ v8f wmma_bf16(v16bf a, v16bf b, v8f c) {
  return __builtin_amdgcn_wmma_f32_16x16x32_bf16(false, a, false, b, (short)0, c, false, false);
}

// ---- f32 -> bf16 (with K padding) ----------------------------------------
__global__ void k_convert_pad(const float* __restrict__ src, bf16_t* __restrict__ dst,
                              int rows, int sk, int dk) {
  long long i = (long long)blockIdx.x * blockDim.x + threadIdx.x;
  long long n = (long long)rows * dk;
  if (i >= n) return;
  int r = (int)(i / dk), k = (int)(i % dk);
  dst[i] = (k < sk) ? f2bf(src[(long long)r * sk + k]) : f2bf(0.0f);
}

// ---- Z = gather(emb, fea) @ W_ih^T + b, stored as Z[t][col][b] ------------
// block: 256 threads (8 waves). Tile: 128 rows x 64 cols. K: 10 x 32, async
// double-buffered LDS staging.
__global__ __launch_bounds__(256)
void k_zgemm(const int* __restrict__ fea, const bf16_t* __restrict__ embbf,
             const bf16_t* __restrict__ wihF, const bf16_t* __restrict__ wihB,
             const float* __restrict__ biasF, const float* __restrict__ biasB,
             bf16_t* __restrict__ ZF, bf16_t* __restrict__ ZB) {
  __shared__ __align__(16) bf16_t As[2][128 * 32];
  __shared__ __align__(16) bf16_t Bs[2][64 * 32];
  const int dir = blockIdx.z;
  const bf16_t* wih = dir ? wihB : wihF;
  const float* bias = dir ? biasB : biasF;
  bf16_t* Z = dir ? ZB : ZF;

  const int tid = threadIdx.x;
  const int w = tid >> 5, lane = tid & 31;
  const int mrow0 = blockIdx.x * 128;
  const int ncol0 = blockIdx.y * 64;

  // two A chunks (16B each; 512 chunks = 4/row) + one B chunk (256 = 4/col)
  const int arow0 = tid >> 2, apart0 = tid & 3;
  const int arow1 = (tid + 256) >> 2, apart1 = (tid + 256) & 3;
  const long long tok0 = fea[mrow0 + arow0];
  const long long tok1 = fea[mrow0 + arow1];
  const int bcol = tid >> 2, bpart = tid & 3;

  auto issue = [&](int kt) {
    const int k0 = kt * 32;
    bf16_t* Ad = As[kt & 1];
    bf16_t* Bd = Bs[kt & 1];
    async_ld_b128(lds_off(Ad + arow0 * 32 + apart0 * 8),
                  embbf + tok0 * DP_ + k0 + apart0 * 8);
    async_ld_b128(lds_off(Ad + arow1 * 32 + apart1 * 8),
                  embbf + tok1 * DP_ + k0 + apart1 * 8);
    async_ld_b128(lds_off(Bd + bcol * 32 + bpart * 8),
                  wih + (long long)(ncol0 + bcol) * DP_ + k0 + bpart * 8);
  };

  const int n = lane & 15, m0 = (lane >> 4) << 3;
  v8f acc[4];
#pragma unroll
  for (int j = 0; j < 4; ++j) {
    float bv = bias[ncol0 + j * 16 + n];
#pragma unroll
    for (int r = 0; r < 8; ++r) acc[j][r] = bv;
  }

  issue(0);
  for (int kt = 0; kt < 10; ++kt) {
    if (kt + 1 < 10) { issue(kt + 1); wait_async3(); }
    else             { wait_async0(); }
    __syncthreads();                       // stage kt resident for all waves
    v16bf a = frag_ab(&As[kt & 1][(w * 16) * 32], 32);
#pragma unroll
    for (int j = 0; j < 4; ++j) {
      v16bf b = frag_ab(&Bs[kt & 1][(j * 16) * 32], 32);
      acc[j] = wmma_bf16(a, b, acc[j]);
    }
    __syncthreads();                       // reads done before buffer reuse
  }
  // store Z[t][col][b] (bf16) so the scan can vector-load 8 consecutive b
#pragma unroll
  for (int j = 0; j < 4; ++j) {
    const int col = ncol0 + j * 16 + n;
#pragma unroll
    for (int r = 0; r < 8; ++r) {
      const int row = mrow0 + w * 16 + m0 + r;
      const int b = row >> 10, t = row & 1023;
      Z[(((long long)t * G4_ + col) << 6) + b] = f2bf(acc[j][r]);
    }
  }
}

// ---- Persistent LSTM scan, batch-split ------------------------------------
// grid = (4 batch-groups, 2 dirs); block = 256 (8 waves). Full W_hh (128KB
// bf16) resident in each WG's LDS; batches independent => no cross-WG sync.
__global__ __launch_bounds__(256)
void k_lstm(const bf16_t* __restrict__ ZF, const bf16_t* __restrict__ ZB,
            const bf16_t* __restrict__ whhF, const bf16_t* __restrict__ whhB,
            const int* __restrict__ length,
            float* __restrict__ hfO, float* __restrict__ hbO,
            bf16_t* __restrict__ lstm_bf) {
  __shared__ __align__(16) bf16_t Wl[G4_ * HD_];   // 131072 B
  __shared__ __align__(16) float  zs[BG_ * G4_];   //  32768 B
  __shared__ __align__(16) bf16_t hl[BG_ * HD_];   //   4096 B
  __shared__ __align__(16) float  cl[BG_ * HD_];   //   8192 B
  __shared__ int lenl[BG_];

  const int grp = blockIdx.x, dir = blockIdx.y;
  const int bbase = grp * BG_;
  const bf16_t* Z = dir ? ZB : ZF;
  const bf16_t* whh = dir ? whhB : whhF;
  float* hOut = dir ? hbO : hfO;

  const int tid = threadIdx.x;
  {  // preload W_hh into LDS (16B vector copies)
    const v8bf* src = (const v8bf*)whh;
    v8bf* dst = (v8bf*)Wl;
    for (int i = tid; i < (G4_ * HD_) / 8; i += 256) dst[i] = src[i];
  }
  for (int i = tid; i < BG_ * HD_; i += 256) { hl[i] = f2bf(0.0f); cl[i] = 0.0f; }
  if (tid < BG_) lenl[tid] = length[bbase + tid];
  __syncthreads();

  const int w = tid >> 5, lane = tid & 31;    // wave w owns cols [w*64, w*64+64)
  const int n = lane & 15, m0 = (lane >> 4) << 3;

  for (int it = 0; it < T_; ++it) {
    const int t = dir ? (T_ - 1 - it) : it;
    // prefetch x-projection for this step (overlaps the WMMA chain below)
    v8bf zv[4];
#pragma unroll
    for (int j = 0; j < 4; ++j) {
      const int col = w * 64 + j * 16 + n;
      zv[j] = *(const v8bf*)(Z + (((long long)t * G4_ + col) << 6) + bbase + m0);
    }
    v8f acc[4];
#pragma unroll
    for (int j = 0; j < 4; ++j)
#pragma unroll
      for (int r = 0; r < 8; ++r) acc[j][r] = 0.0f;
#pragma unroll
    for (int k = 0; k < 4; ++k) {
      v16bf a = frag_ab(&hl[k * 32], HD_);
#pragma unroll
      for (int j = 0; j < 4; ++j) {
        v16bf b = frag_ab(&Wl[(w * 64 + j * 16) * HD_ + k * 32], HD_);
        acc[j] = wmma_bf16(a, b, acc[j]);
      }
    }
#pragma unroll
    for (int j = 0; j < 4; ++j) {
      const int col = w * 64 + j * 16 + n;
#pragma unroll
      for (int r = 0; r < 8; ++r)
        zs[(m0 + r) * G4_ + col] = acc[j][r] + bf2f(zv[j][r]);
    }
    __syncthreads();
    // gates: 16 rows x 128 cols = 2048 pairs over 256 threads; z=[i|f|g|o]
#pragma unroll
    for (int p = 0; p < 8; ++p) {
      const int idx = tid + (p << 8);
      const int lb = idx >> 7, jj = idx & 127;
      const float zi = zs[lb * G4_ + jj];
      const float zf = zs[lb * G4_ + 128 + jj];
      const float zg = zs[lb * G4_ + 256 + jj];
      const float zo = zs[lb * G4_ + 384 + jj];
      float c = sigmoidf_(zf) * cl[lb * HD_ + jj] + sigmoidf_(zi) * tanhf(zg);
      float h = sigmoidf_(zo) * tanhf(c);
      const bool valid = t < lenl[lb];
      const float hm = valid ? h : 0.0f;
      if (dir) { h = hm; c = valid ? c : 0.0f; }  // backward masks recurrence
      hl[lb * HD_ + jj] = f2bf(h);
      cl[lb * HD_ + jj] = c;
      const int bg = bbase + lb;
      hOut[(long long)(bg * T_ + t) * HD_ + jj] = hm;
      lstm_bf[(long long)(bg * T_ + t) * H_ + dir * HD_ + jj] = f2bf(hm);
    }
    __syncthreads();
  }
}

// ---- span mean ------------------------------------------------------------
__global__ void k_tmean(const float* __restrict__ hf, const float* __restrict__ hb,
                        const int* __restrict__ ts, const int* __restrict__ te,
                        float* __restrict__ tmean) {
  const int b = blockIdx.x, h = threadIdx.x;
  const int s = ts[b], e = te[b];
  const float* src = (h < HD_) ? (hf + (long long)b * T_ * HD_ + h)
                               : (hb + (long long)b * T_ * HD_ + (h - HD_));
  float acc = 0.0f;
  for (int t = s; t <= e; ++t) acc += src[(long long)t * HD_];
  tmean[b * H_ + h] = acc / (float)(e - s + 1);
}

// ---- tproj = Wa_t @ t_mean + b_a ------------------------------------------
__global__ void k_tproj(const float* __restrict__ wat, const float* __restrict__ ba,
                        const float* __restrict__ tmean, float* __restrict__ tproj) {
  const int b = blockIdx.x, a = threadIdx.x;
  float acc = ba[a];
  for (int h = 0; h < H_; ++h) acc += wat[a * H_ + h] * tmean[b * H_ + h];
  tproj[b * ATT_ + a] = acc;
}

// ---- fused attention scores: v_a . tanh(lstm @ Wa_h^T + tproj) ------------
__global__ __launch_bounds__(128)
void k_scores(const bf16_t* __restrict__ lstm_bf, const bf16_t* __restrict__ wah,
              const float* __restrict__ tproj, const float* __restrict__ va,
              const int* __restrict__ ts, const int* __restrict__ te,
              float* __restrict__ scores) {
  const int w = threadIdx.x >> 5, lane = threadIdx.x & 31;
  const int r0 = (blockIdx.x * 4 + w) * 16;
  const int n = lane & 15, m0 = (lane >> 4) << 3;
  v8f acc[8];
#pragma unroll
  for (int j = 0; j < 8; ++j)
#pragma unroll
    for (int r = 0; r < 8; ++r)
      acc[j][r] = tproj[((r0 + m0 + r) >> 10) * ATT_ + j * 16 + n];
#pragma unroll
  for (int k = 0; k < 8; ++k) {
    v16bf a = frag_ab(lstm_bf + (long long)r0 * H_ + k * 32, H_);
#pragma unroll
    for (int j = 0; j < 8; ++j) {
      v16bf b = frag_ab(wah + (long long)(j * 16) * H_ + k * 32, H_);
      acc[j] = wmma_bf16(a, b, acc[j]);
    }
  }
  v8f sum;
#pragma unroll
  for (int r = 0; r < 8; ++r) sum[r] = 0.0f;
#pragma unroll
  for (int j = 0; j < 8; ++j) {
    const float vw = va[j * 16 + n];
#pragma unroll
    for (int r = 0; r < 8; ++r) sum[r] += tanhf(acc[j][r]) * vw;
  }
#pragma unroll
  for (int m = 1; m < 16; m <<= 1)
#pragma unroll
    for (int r = 0; r < 8; ++r) sum[r] += __shfl_xor(sum[r], m, 32);
  if (n == 0) {
#pragma unroll
    for (int r = 0; r < 8; ++r) {
      const int row = r0 + m0 + r;
      const int b = row >> 10, t = row & 1023;
      float sc = sum[r];
      if (t >= ts[b] && t <= te[b]) sc = -1.0e9f;
      scores[row] = sc;
    }
  }
}

// ---- softmax over T per batch (in place) ----------------------------------
__global__ void k_softmax(float* __restrict__ scores) {
  __shared__ float red[256];
  const int b = blockIdx.x, tid = threadIdx.x;
  float* sc = scores + b * T_;
  float mx = -1e30f;
  for (int t = tid; t < T_; t += 256) mx = fmaxf(mx, sc[t]);
  red[tid] = mx; __syncthreads();
  for (int s = 128; s > 0; s >>= 1) {
    if (tid < s) red[tid] = fmaxf(red[tid], red[tid + s]);
    __syncthreads();
  }
  mx = red[0]; __syncthreads();
  float sm = 0.0f;
  for (int t = tid; t < T_; t += 256) { float e = __expf(sc[t] - mx); sc[t] = e; sm += e; }
  red[tid] = sm; __syncthreads();
  for (int s = 128; s > 0; s >>= 1) {
    if (tid < s) red[tid] += red[tid + s];
    __syncthreads();
  }
  const float inv = 1.0f / red[0];
  for (int t = tid; t < T_; t += 256) sc[t] *= inv;
}

// ---- context vector s = sum_t alpha * lstm_out ----------------------------
__global__ void k_context(const float* __restrict__ hf, const float* __restrict__ hb,
                          const float* __restrict__ alpha, float* __restrict__ sctx) {
  const int b = blockIdx.x, h = threadIdx.x;
  const float* src = (h < HD_) ? (hf + (long long)b * T_ * HD_ + h)
                               : (hb + (long long)b * T_ * HD_ + (h - HD_));
  const float* al = alpha + b * T_;
  float acc = 0.0f;
  for (int t = 0; t < T_; ++t) acc += al[t] * src[(long long)t * HD_];
  sctx[b * H_ + h] = acc;
}

// ---- output head ----------------------------------------------------------
__global__ void k_out(const float* __restrict__ wout, const float* __restrict__ bout,
                      const float* __restrict__ sctx, float* __restrict__ out) {
  const int i = blockIdx.x * blockDim.x + threadIdx.x;
  if (i >= B_ * CAT_) return;
  const int b = i / CAT_, c = i % CAT_;
  float acc = bout[c];
  for (int h = 0; h < H_; ++h) acc += wout[c * H_ + h] * sctx[b * H_ + h];
  out[i] = acc;
}

extern "C" void kernel_launch(void* const* d_in, const int* in_sizes, int n_in,
                              void* d_out, int out_size, void* d_ws, size_t ws_size,
                              hipStream_t stream) {
  (void)in_sizes; (void)n_in; (void)out_size; (void)ws_size;
  const int*   fea  = (const int*)  d_in[0];
  const int*   len  = (const int*)  d_in[1];
  const int*   ts   = (const int*)  d_in[2];
  const int*   te   = (const int*)  d_in[3];
  const float* emb  = (const float*)d_in[5];
  const float* wihf = (const float*)d_in[6];
  const float* whhf = (const float*)d_in[7];
  const float* bf   = (const float*)d_in[8];
  const float* wihb = (const float*)d_in[9];
  const float* whhb = (const float*)d_in[10];
  const float* bb   = (const float*)d_in[11];
  const float* wah  = (const float*)d_in[12];
  const float* wat  = (const float*)d_in[13];
  const float* ba   = (const float*)d_in[14];
  const float* va   = (const float*)d_in[15];
  const float* wout = (const float*)d_in[16];
  const float* bout = (const float*)d_in[17];

  char* ws = (char*)d_ws;
  size_t o = 0;
  auto alloc = [&](size_t bytes) {
    void* p = ws + o;
    o = (o + bytes + 255) & ~(size_t)255;
    return p;
  };
  bf16_t* embb  = (bf16_t*)alloc((size_t)V_ * DP_ * 2);
  bf16_t* wihfb = (bf16_t*)alloc((size_t)G4_ * DP_ * 2);
  bf16_t* wihbb = (bf16_t*)alloc((size_t)G4_ * DP_ * 2);
  bf16_t* whhfb = (bf16_t*)alloc((size_t)G4_ * HD_ * 2);
  bf16_t* whhbb = (bf16_t*)alloc((size_t)G4_ * HD_ * 2);
  bf16_t* wahb  = (bf16_t*)alloc((size_t)ATT_ * H_ * 2);
  bf16_t* Zf    = (bf16_t*)alloc((size_t)B_ * T_ * G4_ * 2);
  bf16_t* Zb    = (bf16_t*)alloc((size_t)B_ * T_ * G4_ * 2);
  float*  hf    = (float*) alloc((size_t)B_ * T_ * HD_ * 4);
  float*  hb    = (float*) alloc((size_t)B_ * T_ * HD_ * 4);
  bf16_t* lstmb = (bf16_t*)alloc((size_t)B_ * T_ * H_ * 2);
  float*  tmean = (float*) alloc((size_t)B_ * H_ * 4);
  float*  tproj = (float*) alloc((size_t)B_ * ATT_ * 4);
  float*  score = (float*) alloc((size_t)B_ * T_ * 4);
  float*  sctx  = (float*) alloc((size_t)B_ * H_ * 4);

  auto cvt = [&](const float* s, bf16_t* d, int rows, int sk, int dk) {
    long long nn = (long long)rows * dk;
    k_convert_pad<<<(unsigned)((nn + 255) / 256), 256, 0, stream>>>(s, d, rows, sk, dk);
  };
  cvt(emb,  embb,  V_,  D_,  DP_);
  cvt(wihf, wihfb, G4_, D_,  DP_);
  cvt(wihb, wihbb, G4_, D_,  DP_);
  cvt(whhf, whhfb, G4_, HD_, HD_);
  cvt(whhb, whhbb, G4_, HD_, HD_);
  cvt(wah,  wahb,  ATT_, H_, H_);

  k_zgemm<<<dim3(512, 8, 2), 256, 0, stream>>>(fea, embb, wihfb, wihbb, bf, bb, Zf, Zb);
  k_lstm<<<dim3(4, 2), 256, 0, stream>>>(Zf, Zb, whhfb, whhbb, len, hf, hb, lstmb);
  k_tmean<<<B_, H_, 0, stream>>>(hf, hb, ts, te, tmean);
  k_tproj<<<B_, ATT_, 0, stream>>>(wat, ba, tmean, tproj);
  k_scores<<<1024, 128, 0, stream>>>(lstmb, wahb, tproj, va, ts, te, score);
  k_softmax<<<B_, 256, 0, stream>>>(score);
  k_context<<<B_, H_, 0, stream>>>(hf, hb, score, sctx);
  k_out<<<1, 320, 0, stream>>>(wout, bout, sctx, (float*)d_out);
}
